// Interation_44409961841158
// MI455X (gfx1250) — compile-verified, matrix operand order checked by reading, not run on hardware
//
#include <hip/hip_runtime.h>
#include <hip/hip_bf16.h>

typedef __attribute__((ext_vector_type(16))) _Float16 v16h;
typedef __attribute__((ext_vector_type(8)))  _Float16 v8h;
typedef __attribute__((ext_vector_type(8)))  float    v8f;

#define BB   16
#define CC   256
#define NN   2048
#define CQ   64
#define EPSV 1e-5f
#define PSTR 40   // LDS row stride (halfs) for the P tile; multiple of 8 keeps 16B align

__device__ __forceinline__ int kbase_of(int v, int half) {
    // K index of the low element of register-pair v for this lane's half
    return ((v < 4) ? (2 * v) : (16 + 2 * (v - 4))) + 8 * half;
}

// A/B fragment whose K elements are CONTIGUOUS f16 at rowptr[k].
// Per the ISA layout, each lane's data is two contiguous 8-half (16B) runs:
//   f[0..7]  = rowptr[8*half + 0..7]
//   f[8..15] = rowptr[16 + 8*half + 0..7]
__device__ __forceinline__ v16h load_frag_h2(const _Float16* rowptr, int half) {
    v8h lo = *(const v8h*)(rowptr + 8 * half);
    v8h hi = *(const v8h*)(rowptr + 16 + 8 * half);
    v16h f;
#pragma unroll
    for (int i = 0; i < 8; ++i) { f[i] = lo[i]; f[8 + i] = hi[i]; }
    return f;
}

// A/B fragment from f32 memory with element k at base + k*stride (converted to f16)
__device__ __forceinline__ v16h load_frag_f32(const float* base, long stride, int half) {
    v16h f;
#pragma unroll
    for (int v = 0; v < 8; ++v) {
        int kb = kbase_of(v, half);
        f[2 * v]     = (_Float16)base[(long)kb * stride];
        f[2 * v + 1] = (_Float16)base[(long)(kb + 1) * stride];
    }
    return f;
}

__device__ __forceinline__ v8f wmma16(v16h a, v16h b, v8f c) {
    return __builtin_amdgcn_wmma_f32_16x16x32_f16(false, a, false, b, (short)0, c, false, false);
}

// ---------------------------------------------------------------------------
// 1x1 conv + BN + ReLU:  out[o,n] = relu(alpha[o]*(W[o,:].x[:,n]) + delta[o])
// x: [B, C, N] f32, W: [Cout, C] f32.  Output f16, optionally transposed
// to [B, N, Cout] (for q/k so attention fragments are K-contiguous).
// Grid: (N/64, Cout/16, B), block 128 (4 waves, one 16x16 tile each).
// ---------------------------------------------------------------------------
__global__ void proj_bn_relu_kernel(const float* __restrict__ x,
                                    const float* __restrict__ W,
                                    const float* __restrict__ bias,
                                    const float* __restrict__ g,
                                    const float* __restrict__ beta,
                                    const float* __restrict__ mean,
                                    const float* __restrict__ var,
                                    _Float16* __restrict__ out,
                                    int Cout, int transposeOut) {
    const int lane = threadIdx.x & 31;
    const int wave = threadIdx.x >> 5;
    const int half = lane >> 4;
    const int rlow = lane & 15;
    const int b  = blockIdx.z;
    const int o0 = blockIdx.y * 16;
    const int n0 = (blockIdx.x * 4 + wave) * 16;

    const float* xb = x + (size_t)b * CC * NN;
    v8f acc = {};
#pragma unroll
    for (int k0 = 0; k0 < CC; k0 += 32) {
        // A: W rows = output channels, K = input channels (contiguous f32)
        v16h a = load_frag_f32(W + (size_t)(o0 + rlow) * CC + k0, 1, half);
        // B: x, K = input channels (stride N, coalesced across lanes), cols = n
        v16h bf = load_frag_f32(xb + (size_t)k0 * NN + (n0 + rlow), NN, half);
        acc = wmma16(a, bf, acc);
    }
#pragma unroll
    for (int r = 0; r < 8; ++r) {
        int o = o0 + r + 8 * half;
        float alpha = g[o] * rsqrtf(var[o] + EPSV);
        float delta = (bias[o] - mean[o]) * alpha + beta[o];
        float y = fmaxf(acc[r] * alpha + delta, 0.0f);
        _Float16 h = (_Float16)y;
        int n = n0 + rlow;
        if (transposeOut)
            out[((size_t)b * NN + n) * Cout + o] = h;   // [B, N, Cout]
        else
            out[((size_t)b * Cout + o) * NN + n] = h;   // [B, Cout, N]
    }
}

// ---------------------------------------------------------------------------
// Flash-style fused attention: per wave = 16 queries x 128-wide C chunk.
// qT/kT: [B, N, 64] f16, vH: [B, C, N] f16, aoT: [B, N, C] f16.
// Grid: (N/16, B), block 64 (2 waves = the two C halves).
// ---------------------------------------------------------------------------
__global__ void attention_kernel(const _Float16* __restrict__ qT,
                                 const _Float16* __restrict__ kT,
                                 const _Float16* __restrict__ vH,
                                 _Float16* __restrict__ aoT) {
    __shared__ __align__(16) _Float16 plds[2][16 * PSTR];
    const int lane = threadIdx.x & 31;
    const int wave = threadIdx.x >> 5;  // c-half
    const int half = lane >> 4;
    const int rlow = lane & 15;
    const int b  = blockIdx.y;
    const int n0 = blockIdx.x * 16;
    const int c0 = wave * 128;

    // Q A-fragments (rows = queries, K = d in 0..63), loaded once
    const _Float16* qrow = qT + ((size_t)b * NN + n0 + rlow) * CQ;
    v16h qa0 = load_frag_h2(qrow, half);       // d 0..31
    v16h qa1 = load_frag_h2(qrow + 32, half);  // d 32..63

    v8f O[8] = {};
    float mrow[8], lrow[8];
#pragma unroll
    for (int r = 0; r < 8; ++r) { mrow[r] = -__builtin_inff(); lrow[r] = 0.0f; }

    _Float16* P = plds[wave];
    const _Float16* kb0 = kT + (size_t)b * NN * CQ;
    const _Float16* vb0 = vH + (size_t)b * CC * NN;

    for (int m0 = 0; m0 < NN; m0 += 32) {
        // ---- prefetch next m-step (K rows + this wave's V tiles) into cache ----
        if (m0 + 32 < NN) {
            __builtin_prefetch(kb0 + (size_t)(m0 + 32 + rlow + 16 * half) * CQ, 0, 3);
#pragma unroll
            for (int t = 0; t < 8; ++t)
                __builtin_prefetch(vb0 + (size_t)(c0 + t * 16 + rlow) * NN
                                       + m0 + 32 + 16 * half, 0, 3);
        }

        // ---- scores: two 16x16 tiles covering keys m0..m0+31 ----
        const _Float16* krow0 = kb0 + (size_t)(m0 + rlow) * CQ;
        v8f s0 = {}, s1 = {};
        s0 = wmma16(qa0, load_frag_h2(krow0, half), s0);
        s0 = wmma16(qa1, load_frag_h2(krow0 + 32, half), s0);
        s1 = wmma16(qa0, load_frag_h2(krow0 + 16 * CQ, half), s1);
        s1 = wmma16(qa1, load_frag_h2(krow0 + 16 * CQ + 32, half), s1);

        // ---- online softmax (row stats live per (r, lane-half)) ----
#pragma unroll
        for (int r = 0; r < 8; ++r) {
            float mx = fmaxf(s0[r], s1[r]);
            mx = fmaxf(mx, __shfl_xor(mx, 1));
            mx = fmaxf(mx, __shfl_xor(mx, 2));
            mx = fmaxf(mx, __shfl_xor(mx, 4));
            mx = fmaxf(mx, __shfl_xor(mx, 8));
            float mnew  = fmaxf(mrow[r], mx);
            float scale = __expf(mrow[r] - mnew);
            float p0 = __expf(s0[r] - mnew);
            float p1 = __expf(s1[r] - mnew);
            float sum = p0 + p1;
            sum += __shfl_xor(sum, 1);
            sum += __shfl_xor(sum, 2);
            sum += __shfl_xor(sum, 4);
            sum += __shfl_xor(sum, 8);
            lrow[r] = lrow[r] * scale + sum;
            mrow[r] = mnew;
#pragma unroll
            for (int t = 0; t < 8; ++t) O[t][r] *= scale;
            // spill P in D-layout to LDS (lane transpose via LDS)
            int row = r + 8 * half;
            P[row * PSTR + rlow]      = (_Float16)p0;
            P[row * PSTR + 16 + rlow] = (_Float16)p1;
        }
        asm volatile("s_wait_dscnt 0x0" ::: "memory");

        // ---- reload P as an A-fragment (rows = queries, K = 32 keys) ----
        v16h pa;
        {
            const _Float16* prow = P + rlow * PSTR;
            v8h lo = *(const v8h*)(prow + 8 * half);
            v8h hi = *(const v8h*)(prow + 16 + 8 * half);
#pragma unroll
            for (int i = 0; i < 8; ++i) { pa[i] = lo[i]; pa[8 + i] = hi[i]; }
        }

        // ---- O += P @ V^T over this wave's 8 c-tiles ----
#pragma unroll
        for (int t = 0; t < 8; ++t) {
            int c = c0 + t * 16 + rlow;
            v16h vb = load_frag_h2(vb0 + (size_t)c * NN + m0, half);
            O[t] = wmma16(pa, vb, O[t]);
        }
    }

    // ---- normalize and store [B, N, C] f16 ----
#pragma unroll
    for (int r = 0; r < 8; ++r) {
        float inv = 1.0f / lrow[r];
        int n = n0 + r + 8 * half;
#pragma unroll
        for (int t = 0; t < 8; ++t)
            aoT[((size_t)b * NN + n) * CC + c0 + t * 16 + rlow] =
                (_Float16)(O[t][r] * inv);
    }
}

// ---------------------------------------------------------------------------
// Final conv + BN + ReLU + residual: out = n3 + gamma * relu(bn(Wc @ ao))
// aoT: [B, N, C] f16; n3/out: [B, C, N] f32.
// Grid: (N/64, C/16, B), block 128.
// ---------------------------------------------------------------------------
__global__ void final_conv_residual_kernel(const _Float16* __restrict__ aoT,
                                           const float* __restrict__ Wc,
                                           const float* __restrict__ bias,
                                           const float* __restrict__ g,
                                           const float* __restrict__ beta,
                                           const float* __restrict__ mean,
                                           const float* __restrict__ var,
                                           const float* __restrict__ n3,
                                           const float* __restrict__ gamma,
                                           float* __restrict__ out) {
    const int lane = threadIdx.x & 31;
    const int wave = threadIdx.x >> 5;
    const int half = lane >> 4;
    const int rlow = lane & 15;
    const int b  = blockIdx.z;
    const int o0 = blockIdx.y * 16;
    const int n0 = (blockIdx.x * 4 + wave) * 16;

    v8f acc = {};
#pragma unroll
    for (int k0 = 0; k0 < CC; k0 += 32) {
        v16h a  = load_frag_f32(Wc + (size_t)(o0 + rlow) * CC + k0, 1, half);
        v16h bf = load_frag_h2(aoT + ((size_t)b * NN + n0 + rlow) * CC + k0, half);
        acc = wmma16(a, bf, acc);
    }
    const float gm = gamma[0];
#pragma unroll
    for (int r = 0; r < 8; ++r) {
        int o = o0 + r + 8 * half;
        float alpha = g[o] * rsqrtf(var[o] + EPSV);
        float delta = (bias[o] - mean[o]) * alpha + beta[o];
        float y = fmaxf(acc[r] * alpha + delta, 0.0f);
        size_t idx = ((size_t)b * CC + o) * NN + n0 + rlow;
        out[idx] = n3[idx] + gm * y;
    }
}

extern "C" void kernel_launch(void* const* d_in, const int* in_sizes, int n_in,
                              void* d_out, int out_size, void* d_ws, size_t ws_size,
                              hipStream_t stream) {
    const float* n1 = (const float*)d_in[0];
    const float* n2 = (const float*)d_in[1];
    const float* n3 = (const float*)d_in[2];
    const float* Wq = (const float*)d_in[3];
    const float* bq = (const float*)d_in[4];
    const float* gq = (const float*)d_in[5];
    const float* betaq = (const float*)d_in[6];
    const float* mq = (const float*)d_in[7];
    const float* vq = (const float*)d_in[8];
    const float* Wk = (const float*)d_in[9];
    const float* bk = (const float*)d_in[10];
    const float* gk = (const float*)d_in[11];
    const float* betak = (const float*)d_in[12];
    const float* mk = (const float*)d_in[13];
    const float* vk = (const float*)d_in[14];
    const float* Wv = (const float*)d_in[15];
    const float* bv = (const float*)d_in[16];
    const float* gv = (const float*)d_in[17];
    const float* betav = (const float*)d_in[18];
    const float* mv = (const float*)d_in[19];
    const float* vv = (const float*)d_in[20];
    const float* Wc = (const float*)d_in[21];
    const float* bc = (const float*)d_in[22];
    const float* gc = (const float*)d_in[23];
    const float* betac = (const float*)d_in[24];
    const float* mc = (const float*)d_in[25];
    const float* vc = (const float*)d_in[26];
    const float* gamma = (const float*)d_in[27];
    float* out = (float*)d_out;

    // workspace: qT [B,N,64] + kT [B,N,64] + vH [B,C,N] + aoT [B,N,C], all f16
    _Float16* qT  = (_Float16*)d_ws;
    _Float16* kT  = qT + (size_t)BB * NN * CQ;
    _Float16* vH  = kT + (size_t)BB * NN * CQ;
    _Float16* aoT = vH + (size_t)BB * CC * NN;

    dim3 blk(128, 1, 1);
    dim3 grdQK(NN / 64, CQ / 16, BB);
    dim3 grdV(NN / 64, CC / 16, BB);

    proj_bn_relu_kernel<<<grdQK, blk, 0, stream>>>(n1, Wq, bq, gq, betaq, mq, vq, qT, CQ, 1);
    proj_bn_relu_kernel<<<grdQK, blk, 0, stream>>>(n2, Wk, bk, gk, betak, mk, vk, kT, CQ, 1);
    proj_bn_relu_kernel<<<grdV,  blk, 0, stream>>>(n3, Wv, bv, gv, betav, mv, vv, vH, CC, 0);

    dim3 attBlk(64, 1, 1);
    dim3 attGrd(NN / 16, BB, 1);
    attention_kernel<<<attGrd, attBlk, 0, stream>>>(qT, kT, vH, aoT);

    final_conv_residual_kernel<<<grdV, blk, 0, stream>>>(aoT, Wc, bc, gc, betac, mc, vc,
                                                         n3, gamma, out);
}
